// GatedDeltaNet_6614249636542
// MI455X (gfx1250) — matrix operand order repaired and checked by reading, not compiled, not run on hardware
//
#include <hip/hip_runtime.h>
#include <hip/hip_bf16.h>
#include <math.h>

typedef __bf16 bf16;
typedef __attribute__((ext_vector_type(16))) __bf16 v16bf;
typedef __attribute__((ext_vector_type(8)))  float  v8f;

#define WMMA_BF16(A,B,C) __builtin_amdgcn_wmma_f32_16x16x32_bf16(false,(A),false,(B),(short)0,(C),false,false)

constexpr int B_  = 2;
constexpr int T_  = 2048;
constexpr int H_  = 6;
constexpr int DK_ = 256;
constexpr int DV_ = 256;
constexpr int HID_= 2048;
constexpr int CH  = 16;          // chunk length (time tile)
constexpr int NCH = T_ / CH;     // 128 chunks

__device__ inline float wred(float x) {
#pragma unroll
    for (int m = 16; m >= 1; m >>= 1) x += __shfl_xor(x, m, 32);
    return x;
}

__device__ inline float dot8(const float4& a, const float4& b,
                             const float4& c, const float4& d) {
    return a.x*b.x + a.y*b.y + a.z*b.z + a.w*b.w
         + c.x*d.x + c.y*d.y + c.z*d.z + c.w*d.w;
}

// pack two floats into a dword of two bf16 (v_cvt_pk_bf16_f32)
__device__ inline unsigned int pkbf(float a, float b) {
    union { struct { bf16 x, y; } s; unsigned int u; } t;
    t.s.x = (bf16)a; t.s.y = (bf16)b;
    return t.u;
}

union V32 { v16bf v; uint4 u[2]; };

// ---- WMMA operand loaders: contiguous 16B/32B LDS reads ----
// A-matrix 16x32 bf16: lane = m + 16*((kk/8)%2); per lane two contiguous 8-elem runs
__device__ inline v16bf loadA(const bf16* src, int ld, int k0, int lane) {
    int m  = lane & 15;
    int kb = (lane < 16) ? 0 : 8;
    const bf16* p = src + m*ld + k0 + kb;
    V32 t;
    t.u[0] = *(const uint4*)(p);
    t.u[1] = *(const uint4*)(p + 16);
    return t.v;
}
// B-matrix 32x16 bf16 from n-major source srcT[n][k]: lane = n + 16*((kk/8)%2); per lane two contiguous 8-elem runs
__device__ inline v16bf loadB(const bf16* srcT, int ld, int k0, int lane) {
    int n  = lane & 15;
    int kb = (lane < 16) ? 0 : 8;
    const bf16* p = srcT + n*ld + k0 + kb;
    V32 t;
    t.u[0] = *(const uint4*)(p);
    t.u[1] = *(const uint4*)(p + 16);
    return t.v;
}

// ================= Kernel 1: projections (beta, g, gate) ==================
__global__ void __launch_bounds__(256)
gdn_proj_kernel(const float* __restrict__ hab, const float* __restrict__ hg,
                const float* __restrict__ b_w, const float* __restrict__ a_w,
                const float* __restrict__ g_w, const float* __restrict__ A_log,
                const float* __restrict__ dt_bias,
                float* __restrict__ beta_o, float* __restrict__ g_o,
                float* __restrict__ gate_o)
{
    int wid  = (blockIdx.x * blockDim.x + threadIdx.x) >> 5;  // (b*T + t)
    int lane = threadIdx.x & 31;
    if (wid >= B_ * T_) return;
    const float* ab = hab + (size_t)wid * HID_;
    const float* hgp = hg + (size_t)wid * HID_;
    for (int h = 0; h < H_; ++h) {
        const float* bw = b_w + h * HID_;
        const float* aw = a_w + h * HID_;
        const float* gw = g_w + h * HID_;
        float xb = 0.f, xa = 0.f, xg = 0.f;
        for (int i = lane * 4; i < HID_; i += 128) {
            float4 a4 = *(const float4*)(ab + i);
            float4 g4 = *(const float4*)(hgp + i);
            float4 wb = *(const float4*)(bw + i);
            float4 wa = *(const float4*)(aw + i);
            float4 wg = *(const float4*)(gw + i);
            xb += a4.x*wb.x + a4.y*wb.y + a4.z*wb.z + a4.w*wb.w;
            xa += a4.x*wa.x + a4.y*wa.y + a4.z*wa.z + a4.w*wa.w;
            xg += g4.x*wg.x + g4.y*wg.y + g4.z*wg.z + g4.w*wg.w;
        }
        xb = wred(xb); xa = wred(xa); xg = wred(xg);
        if (lane == 0) {
            int o = wid * H_ + h;
            beta_o[o] = 1.f / (1.f + __expf(-xb));
            float sp_in = xa + dt_bias[h];
            float sp = (sp_in > 20.f) ? sp_in : log1pf(__expf(sp_in));
            g_o[o]    = -__expf(A_log[h]) * sp;
            gate_o[o] = xg;
        }
    }
}

// ================= Kernel 2: chunked gated delta rule =====================
struct __align__(32) SmemMain {
    bf16  Kn [CH][DK_];    // normalized k rows (A-layout src AND K^T B-operand src)
    bf16  KbA[CH][DK_];    // beta * k
    bf16  KdA[CH][DK_];    // beta * e^L * k
    bf16  QA_[CH][DK_];    // q * dk^-1/2
    bf16  QdA[CH][DK_];    // e^L * q * dk^-1/2
    bf16  SbT[16][DK_];    // state snapshot, n-major (B operand)
    bf16  KeT[DK_][32];    // e^(L15-Lt)*k_t[kcol], t contiguous, cols 16..31 = 0
    bf16  DbT[16][32];     // Delta bf16, n-major, cols 16..31 = 0 (B operand)
    bf16  Aqb[CH][32];     // decayed Q@K^T lower-tri bf16, cols 16..31 = 0 (A operand)
    float gl[CH], Ll[CH], bl[CH], el[CH], fe[CH];
    float Af[CH][CH];      // strictly-lower decayed A (f32, for solver)
    float Rf[CH][CH];      // beta*v - KS
};

__global__ void __launch_bounds__(32)
gdn_main_kernel(const float* __restrict__ qg, const float* __restrict__ kg,
                const float* __restrict__ vg, const float* __restrict__ gbuf,
                const float* __restrict__ bbuf, float* __restrict__ obuf)
{
    __shared__ SmemMain sm;
    const int lane = threadIdx.x;
    const int vb = blockIdx.x & 15;          // v-column block (16 cols)
    const int bh = blockIdx.x >> 4;
    const int b = bh / H_, h = bh % H_;
    const int v0 = vb * 16;
    const float rdk = 0.0625f;               // DK^-0.5

    v8f S[16];
#pragma unroll
    for (int i = 0; i < 16; ++i) S[i] = v8f{};

    const int n  = lane & 15;                // C/D tile column for this lane
    const int hi = lane >> 4;

    // ---- one-time zeroing of operand pad regions (cols 16..31) ----
    for (int i = lane; i < DK_; i += 32) {
        *(uint4*)&sm.KeT[i][16] = uint4{0,0,0,0};
        *(uint4*)&sm.KeT[i][24] = uint4{0,0,0,0};
    }
    if (lane < 16) {
        *(uint4*)&sm.DbT[lane][16] = uint4{0,0,0,0};
        *(uint4*)&sm.DbT[lane][24] = uint4{0,0,0,0};
        *(uint4*)&sm.Aqb[lane][16] = uint4{0,0,0,0};
        *(uint4*)&sm.Aqb[lane][24] = uint4{0,0,0,0};
    }

    for (int c = 0; c < NCH; ++c) {
        const int t0 = c * CH;
        // ---- g, beta; prefix L; scale tables ----
        if (lane < CH) {
            int gi = (b*T_ + t0 + lane) * H_ + h;
            sm.gl[lane] = gbuf[gi];
            sm.bl[lane] = bbuf[gi];
        }
        if (lane < CH) {
            float acc = 0.f;
            for (int j = 0; j <= lane; ++j) acc += sm.gl[j];
            sm.Ll[lane] = acc;
            sm.el[lane] = __expf(acc);
        }
        const float Lend = sm.Ll[15];
        if (lane < CH) sm.fe[lane] = __expf(Lend - sm.Ll[lane]);

        // ---- load + l2norm q,k rows; write pre-scaled bf16 copies ----
        for (int r = 0; r < CH; ++r) {
            size_t base = ((size_t)(b*T_ + t0 + r) * H_ + h) * DK_;
            float4 k1 = *(const float4*)(kg + base + lane*8);
            float4 k2 = *(const float4*)(kg + base + lane*8 + 4);
            float4 q1 = *(const float4*)(qg + base + lane*8);
            float4 q2 = *(const float4*)(qg + base + lane*8 + 4);
            float rk = rsqrtf(wred(dot8(k1, k1, k2, k2)) + 1e-6f);
            float rq = rsqrtf(wred(dot8(q1, q1, q2, q2)) + 1e-6f) * rdk;
            float kn[8] = { k1.x*rk, k1.y*rk, k1.z*rk, k1.w*rk,
                            k2.x*rk, k2.y*rk, k2.z*rk, k2.w*rk };
            float qn[8] = { q1.x*rq, q1.y*rq, q1.z*rq, q1.w*rq,
                            q2.x*rq, q2.y*rq, q2.z*rq, q2.w*rq };
            float be = sm.bl[r], eL = sm.el[r], fer = sm.fe[r];
            float bd = be * eL;
            int c0 = lane * 8;
            *(uint4*)&sm.Kn [r][c0] = uint4{ pkbf(kn[0],kn[1]), pkbf(kn[2],kn[3]),
                                             pkbf(kn[4],kn[5]), pkbf(kn[6],kn[7]) };
            *(uint4*)&sm.KbA[r][c0] = uint4{ pkbf(be*kn[0],be*kn[1]), pkbf(be*kn[2],be*kn[3]),
                                             pkbf(be*kn[4],be*kn[5]), pkbf(be*kn[6],be*kn[7]) };
            *(uint4*)&sm.KdA[r][c0] = uint4{ pkbf(bd*kn[0],bd*kn[1]), pkbf(bd*kn[2],bd*kn[3]),
                                             pkbf(bd*kn[4],bd*kn[5]), pkbf(bd*kn[6],bd*kn[7]) };
            *(uint4*)&sm.QA_[r][c0] = uint4{ pkbf(qn[0],qn[1]), pkbf(qn[2],qn[3]),
                                             pkbf(qn[4],qn[5]), pkbf(qn[6],qn[7]) };
            *(uint4*)&sm.QdA[r][c0] = uint4{ pkbf(eL*qn[0],eL*qn[1]), pkbf(eL*qn[2],eL*qn[3]),
                                             pkbf(eL*qn[4],eL*qn[5]), pkbf(eL*qn[6],eL*qn[7]) };
#pragma unroll
            for (int j = 0; j < 8; ++j)
                sm.KeT[c0 + j][r] = (bf16)(fer * kn[j]);
        }
        // ---- prefetch next chunk's q/k ----
        if (c + 1 < NCH) {
            size_t nb = ((size_t)(b*T_ + t0 + CH + (lane & 15)) * H_ + h) * DK_ + (lane >> 4) * 128;
            __builtin_prefetch(kg + nb, 0, 1);
            __builtin_prefetch(qg + nb, 0, 1);
        }
        // ---- snapshot state S -> SbT (n-major bf16), packed stores ----
#pragma unroll
        for (int kt = 0; kt < 16; ++kt) {
            *(uint4*)&sm.SbT[n][kt*16 + 8*hi] =
                uint4{ pkbf(S[kt][0], S[kt][1]), pkbf(S[kt][2], S[kt][3]),
                       pkbf(S[kt][4], S[kt][5]), pkbf(S[kt][6], S[kt][7]) };
        }
        // ---- four 16x16x256 products via bf16 WMMA ----
        v8f accKS = v8f{}, accOi = v8f{}, accA = v8f{}, accAq = v8f{};
        for (int s = 0; s < 8; ++s) {
            int k0 = s * 32;
            v16bf SB  = loadB(&sm.SbT[0][0], DK_, k0, lane);
            v16bf KT  = loadB(&sm.Kn [0][0], DK_, k0, lane);
            v16bf Akd = loadA(&sm.KdA[0][0], DK_, k0, lane);
            v16bf Akb = loadA(&sm.KbA[0][0], DK_, k0, lane);
            v16bf Aqd = loadA(&sm.QdA[0][0], DK_, k0, lane);
            v16bf Aq1 = loadA(&sm.QA_[0][0], DK_, k0, lane);
            accKS = WMMA_BF16(Akd, SB, accKS);   // beta*e^L * (K @ S)
            accOi = WMMA_BF16(Aqd, SB, accOi);   // e^L * (Q @ S)
            accA  = WMMA_BF16(Akb, KT, accA);    // beta * (K @ K^T)
            accAq = WMMA_BF16(Aq1, KT, accAq);   // Q @ K^T
        }
        // ---- decay masks, R = beta*v - KS ----
#pragma unroll
        for (int r8 = 0; r8 < 8; ++r8) {
            int m = r8 + 8*hi;                    // time row i
            float dec = __expf(sm.Ll[m] - sm.Ll[n]);
            sm.Af[m][n] = (n < m)  ? accA[r8] * dec : 0.f;
            sm.Aqb[m][n] = (bf16)((n <= m) ? accAq[r8] * dec : 0.f);
            float vv = vg[((size_t)(b*T_ + t0 + m) * H_ + h) * DV_ + v0 + n];
            sm.Rf[m][n] = sm.bl[m] * vv - accKS[r8];
        }
        // ---- forward substitution: (I+A) Delta = R (lane = column, Delta in regs) ----
        if (lane < 16) {
            float Dloc[16];
#pragma unroll
            for (int i = 0; i < 16; ++i) {
                float d = sm.Rf[i][lane];
#pragma unroll
                for (int j = 0; j < i; ++j) d -= sm.Af[i][j] * Dloc[j];
                Dloc[i] = d;
                sm.DbT[lane][i] = (bf16)d;
            }
        }
        // ---- B operand: Delta zero-padded [32x16] (pad comes from LDS zeros) ----
        v16bf DB = loadB(&sm.DbT[0][0], 32, 0, lane);
        // ---- output: o = Aq @ Delta + e^L Q S  (one padded WMMA) ----
        v8f accO = WMMA_BF16(loadA(&sm.Aqb[0][0], 32, 0, lane), DB, accOi);
#pragma unroll
        for (int r8 = 0; r8 < 8; ++r8) {
            int m = r8 + 8*hi;
            obuf[((size_t)(b*T_ + t0 + m) * H_ + h) * DV_ + v0 + n] = accO[r8];
        }
        // ---- state update: S = e^Lend * S + Kdec^T @ Delta ----
        const float eLend = __expf(Lend);
#pragma unroll
        for (int kt = 0; kt < 16; ++kt) {
#pragma unroll
            for (int r8 = 0; r8 < 8; ++r8) S[kt][r8] *= eLend;
            S[kt] = WMMA_BF16(loadA(&sm.KeT[kt*16][0], 32, 0, lane), DB, S[kt]);
        }
    }
}

// ========== Kernel 3: RMSNorm + swish gate + per-head output GEMM =========
struct __align__(32) SmemOut {
    bf16  On [CH][DV_];    // normalized, gated o rows (A operand src)
    bf16  WtT[16][DV_];    // o_w tile transposed: [d][v] (B operand src)
    float sw[CH];
};

__global__ void __launch_bounds__(32)
gdn_out_kernel(const float* __restrict__ obuf, const float* __restrict__ gatebuf,
               const float* __restrict__ norm_w, const float* __restrict__ o_w,
               float* __restrict__ out)
{
    __shared__ SmemOut sm;
    const int lane = threadIdx.x;
    const int tt = blockIdx.x & (NCH - 1);
    const int bh = blockIdx.x >> 7;
    const int b = bh / H_, h = bh % H_;
    const int t0 = tt * CH;

    if (lane < CH) {
        float gt = gatebuf[(b*T_ + t0 + lane) * H_ + h];
        sm.sw[lane] = gt / (1.f + __expf(-gt));
    }
    for (int r = 0; r < CH; ++r) {
        size_t base = ((size_t)(b*T_ + t0 + r) * H_ + h) * DV_;
        float4 o1 = *(const float4*)(obuf + base + lane*8);
        float4 o2 = *(const float4*)(obuf + base + lane*8 + 4);
        float s = rsqrtf(wred(dot8(o1, o1, o2, o2)) * (1.f / DV_) + 1e-5f) * sm.sw[r];
        float4 w1 = *(const float4*)(norm_w + lane*8);
        float4 w2 = *(const float4*)(norm_w + lane*8 + 4);
        *(uint4*)&sm.On[r][lane*8] =
            uint4{ pkbf(o1.x*w1.x*s, o1.y*w1.y*s), pkbf(o1.z*w1.z*s, o1.w*w1.w*s),
                   pkbf(o2.x*w2.x*s, o2.y*w2.y*s), pkbf(o2.z*w2.z*s, o2.w*w2.w*s) };
    }
    // cache the 8 A operands for this 16x256 activation tile
    v16bf AOn[8];
#pragma unroll
    for (int s = 0; s < 8; ++s) AOn[s] = loadA(&sm.On[0][0], DV_, s*32, lane);

    const float* W = o_w + (size_t)h * DV_ * DK_;
    for (int dt = 0; dt < 16; ++dt) {
        int d0 = dt * 16;
        // load o_w[:, d0:d0+16) -> WtT[d][v] (transposed, bf16)
        for (int i4 = lane*4; i4 < DV_*16; i4 += 128) {
            int vv = i4 >> 4, dd = i4 & 15;
            float4 w4 = *(const float4*)(W + (size_t)vv * DK_ + d0 + dd);
            sm.WtT[dd  ][vv] = (bf16)w4.x;
            sm.WtT[dd+1][vv] = (bf16)w4.y;
            sm.WtT[dd+2][vv] = (bf16)w4.z;
            sm.WtT[dd+3][vv] = (bf16)w4.w;
        }
        v8f acc = v8f{};
        for (int s = 0; s < 8; ++s) {
            v16bf WB = loadB(&sm.WtT[0][0], DV_, s*32, lane);
            acc = WMMA_BF16(AOn[s], WB, acc);
        }
#pragma unroll
        for (int r8 = 0; r8 < 8; ++r8) {
            int m = r8 + 8*(lane >> 4);
            out[(size_t)(b*T_ + t0 + m) * (H_*DK_) + h*DK_ + d0 + (lane & 15)] = acc[r8];
        }
    }
}

// =========================== launcher =====================================
extern "C" void kernel_launch(void* const* d_in, const int* in_sizes, int n_in,
                              void* d_out, int out_size, void* d_ws, size_t ws_size,
                              hipStream_t stream) {
    const float* hab    = (const float*)d_in[0];
    const float* hg     = (const float*)d_in[1];
    const float* q      = (const float*)d_in[2];
    const float* k      = (const float*)d_in[3];
    const float* v      = (const float*)d_in[4];
    const float* b_w    = (const float*)d_in[5];
    const float* a_w    = (const float*)d_in[6];
    const float* A_log  = (const float*)d_in[7];
    const float* dt_bias= (const float*)d_in[8];
    const float* g_w    = (const float*)d_in[9];
    const float* norm_w = (const float*)d_in[10];
    const float* o_w    = (const float*)d_in[11];

    float* beta = (float*)d_ws;                    // [B,T,H]
    float* g    = beta + B_*T_*H_;                 // [B,T,H]
    float* gate = g    + B_*T_*H_;                 // [B,T,H]
    float* obuf = gate + B_*T_*H_;                 // [B,T,H,DV]

    gdn_proj_kernel<<<dim3((B_*T_)/8), dim3(256), 0, stream>>>(
        hab, hg, b_w, a_w, g_w, A_log, dt_bias, beta, g, gate);

    gdn_main_kernel<<<dim3(B_*H_*(DV_/16)), dim3(32), 0, stream>>>(
        q, k, v, g, beta, obuf);

    gdn_out_kernel<<<dim3(B_*H_*NCH), dim3(32), 0, stream>>>(
        obuf, gate, norm_w, o_w, (float*)d_out);
}
